// BFPLinear_180388626581
// MI455X (gfx1250) — 8-bit matrix operand order repaired and checked by reading, compiled, not run on hardware
//
#include <hip/hip_runtime.h>


typedef __attribute__((ext_vector_type(8))) int v8i;

#define DIM  4096
#define GRP  64
#define NGRP (DIM / GRP)
#define ROWPAD 80
#define A_STG (64 * ROWPAD)
#define B_STG (256 * ROWPAD)

__global__ __launch_bounds__(256) void bfp_quantize_kernel(
    const float* __restrict__ in, signed char* __restrict__ q,
    float* __restrict__ scale_t) {
  const int wave = (blockIdx.x * blockDim.x + threadIdx.x) >> 5;
  const int lane = threadIdx.x & 31;
  const int row  = wave >> 6;
  const int g    = wave & (NGRP - 1);

  const size_t base = (size_t)row * DIM + (size_t)g * GRP + (size_t)lane * 2;
  const float2 v = *(const float2*)(in + base);

  float m = fmaxf(fabsf(v.x), fabsf(v.y));
#pragma unroll
  for (int off = 16; off >= 1; off >>= 1)
    m = fmaxf(m, __shfl_xor(m, off, 32));

  const float e    = floorf(log2f(fmaxf(m, 1e-38f)));
  const float inv  = exp2f(7.0f - e);
  const float step = exp2f(e - 7.0f);

  const float qa = (m == 0.0f) ? 0.0f : fminf(fmaxf(rintf(v.x * inv), -128.0f), 127.0f);
  const float qb = (m == 0.0f) ? 0.0f : fminf(fmaxf(rintf(v.y * inv), -128.0f), 127.0f);
  const int ia = (int)qa, ib = (int)qb;
  *(unsigned short*)(q + base) =
      (unsigned short)((ia & 0xFF) | ((ib & 0xFF) << 8));

  if (lane == 0) scale_t[(size_t)g * DIM + row] = step;
}

__device__ __forceinline__ void async_copy_b128(unsigned lds_addr, const void* gptr) {
  const unsigned long long ga = (unsigned long long)gptr;
  asm volatile("global_load_async_to_lds_b128 %0, %1, off"
               :: "v"(lds_addr), "v"(ga) : "memory");
}

__global__ __launch_bounds__(256) void bfp_gemm_iu8_wmma(
    const signed char* __restrict__ xq, const signed char* __restrict__ wq,
    const float* __restrict__ sx_t, const float* __restrict__ sw_t,
    const float* __restrict__ bias, float* __restrict__ out) {
  __shared__ signed char sA[3 * A_STG];
  __shared__ signed char sB[3 * B_STG];

  const int t     = threadIdx.x;
  const int lane  = t & 31;
  const int wave  = t >> 5;
  const int waveM = wave >> 2;
  const int waveN = wave & 3;
  const int blkM  = blockIdx.x >> 4;
  const int blkN  = blockIdx.x & 15;
  const int baseM = blkM * 64;
  const int baseN = blkN * 256;
  const int half  = lane >> 4;
  const int l16   = lane & 15;

  const int srow   = t >> 2;
  const int schunk = (t & 3) * 16;
  const signed char* aG = xq + (size_t)(baseM + srow) * DIM + schunk;
  const signed char* bG = wq + (size_t)(baseN + srow) * DIM + schunk;
  const unsigned aLbase = (unsigned)(size_t)(&sA[0]) + srow * ROWPAD + schunk;
  const unsigned bLbase = (unsigned)(size_t)(&sB[0]) + srow * ROWPAD + schunk;

  unsigned aRd[2], bRd[4];
#pragma unroll
  for (int mi = 0; mi < 2; ++mi)
    aRd[mi] = (unsigned)((waveM * 32 + mi * 16 + l16) * ROWPAD + half * 16);
#pragma unroll
  for (int ni = 0; ni < 4; ++ni)
    bRd[ni] = (unsigned)((waveN * 64 + ni * 16 + l16) * ROWPAD + half * 16);

  const float* sxPtr[2];
  const float* swPtr[4];
#pragma unroll
  for (int mi = 0; mi < 2; ++mi)
    sxPtr[mi] = sx_t + baseM + waveM * 32 + mi * 16 + half * 8;
#pragma unroll
  for (int ni = 0; ni < 4; ++ni)
    swPtr[ni] = sw_t + baseN + waveN * 64 + ni * 16 + l16;

  auto stage = [&](int g, int s) {
    const size_t goff = (size_t)g * GRP;
    async_copy_b128(aLbase + s * A_STG, aG + goff);
#pragma unroll
    for (int j = 0; j < 4; ++j)
      async_copy_b128(bLbase + s * B_STG + j * 64 * ROWPAD,
                      bG + goff + (size_t)j * 64 * DIM);
  };

  float acc[2][4][8] = {};
  const v8i czero = {0, 0, 0, 0, 0, 0, 0, 0};

  stage(0, 0);
  stage(1, 1);

  for (int g = 0; g < NGRP; ++g) {
    const int s = g % 3;
    if (g + 1 < NGRP) asm volatile("s_wait_asynccnt 0x5" ::: "memory");
    else              asm volatile("s_wait_asynccnt 0x0" ::: "memory");
    __syncthreads();
    if (g + 2 < NGRP) stage(g + 2, (g + 2) % 3);

    v8i A[2], B[4];
#pragma unroll
    for (int mi = 0; mi < 2; ++mi) {
      const signed char* p = sA + s * A_STG + aRd[mi];
      const int4 a0 = *(const int4*)(p);
      const int4 a1 = *(const int4*)(p + 32);
      A[mi] = (v8i){a0.x, a0.y, a0.z, a0.w, a1.x, a1.y, a1.z, a1.w};
    }
#pragma unroll
    for (int ni = 0; ni < 4; ++ni) {
      const signed char* p = sB + s * B_STG + bRd[ni];
      const int4 b0 = *(const int4*)(p);
      const int4 b1 = *(const int4*)(p + 32);
      B[ni] = (v8i){b0.x, b0.y, b0.z, b0.w, b1.x, b1.y, b1.z, b1.w};
    }

    float4 sx0[2], sx1[2];
    float  swv[4];
#pragma unroll
    for (int mi = 0; mi < 2; ++mi) {
      sx0[mi] = *(const float4*)(sxPtr[mi] + (size_t)g * DIM);
      sx1[mi] = *(const float4*)(sxPtr[mi] + (size_t)g * DIM + 4);
    }
#pragma unroll
    for (int ni = 0; ni < 4; ++ni) swv[ni] = swPtr[ni][(size_t)g * DIM];

#pragma unroll
    for (int mi = 0; mi < 2; ++mi) {
#pragma unroll
      for (int ni = 0; ni < 4; ++ni) {
        const v8i Ci = __builtin_amdgcn_wmma_i32_16x16x64_iu8(
             true, A[mi],  true, B[ni], czero,
             false,  false);
        const float sc = swv[ni];
        acc[mi][ni][0] += (float)Ci[0] * (sx0[mi].x * sc);
        acc[mi][ni][1] += (float)Ci[1] * (sx0[mi].y * sc);
        acc[mi][ni][2] += (float)Ci[2] * (sx0[mi].z * sc);
        acc[mi][ni][3] += (float)Ci[3] * (sx0[mi].w * sc);
        acc[mi][ni][4] += (float)Ci[4] * (sx1[mi].x * sc);
        acc[mi][ni][5] += (float)Ci[5] * (sx1[mi].y * sc);
        acc[mi][ni][6] += (float)Ci[6] * (sx1[mi].z * sc);
        acc[mi][ni][7] += (float)Ci[7] * (sx1[mi].w * sc);
      }
    }
  }

#pragma unroll
  for (int ni = 0; ni < 4; ++ni) {
    const float bv = 2.0f * bias[baseN + waveN * 64 + ni * 16 + l16];
#pragma unroll
    for (int mi = 0; mi < 2; ++mi) {
      float* op = out + (size_t)(baseM + waveM * 32 + mi * 16 + half * 8) * DIM
                      + baseN + waveN * 64 + ni * 16 + l16;
#pragma unroll
      for (int r = 0; r < 8; ++r) op[(size_t)r * DIM] = acc[mi][ni][r] + bv;
    }
  }
}

extern "C" void kernel_launch(void* const* d_in, const int* in_sizes, int n_in,
                              void* d_out, int out_size, void* d_ws, size_t ws_size,
                              hipStream_t stream) {
  const float* x    = (const float*)d_in[0];
  const float* w    = (const float*)d_in[1];
  const float* bias = (const float*)d_in[2];
  float* out = (float*)d_out;

  char* ws = (char*)d_ws;
  signed char* xq = (signed char*)ws;
  signed char* wq = (signed char*)(ws + (size_t)16 * 1024 * 1024);
  float* sx_t = (float*)(ws + (size_t)32 * 1024 * 1024);
  float* sw_t = (float*)(ws + (size_t)33 * 1024 * 1024);

  const int qblocks = (DIM * NGRP) / 8;
  bfp_quantize_kernel<<<qblocks, 256, 0, stream>>>(x, xq, sx_t);
  bfp_quantize_kernel<<<qblocks, 256, 0, stream>>>(w, wq, sw_t);

  bfp_gemm_iu8_wmma<<<64 * 16, 256, 0, stream>>>(xq, wq, sx_t, sw_t, bias, out);
}
